// TLSTM_52664888983986
// MI455X (gfx1250) — compile-verified, matrix operand order checked by reading, not run on hardware
//
#include <hip/hip_runtime.h>

typedef unsigned short ushort_t;
typedef __attribute__((ext_vector_type(16))) __bf16 v16bf;
typedef __attribute__((ext_vector_type(8)))  float  v8f;

#define Tn 128
#define Bn 64
#define Fn 512
#define Hn 1024
#define Cn 513

#define CHUNK   128          // K-values staged per LDS chunk
#define LDSPAD  8            // bf16 elems of row padding (16B) to kill bank conflicts
#define NCHUNK  12           // 4 chunks of x (K=512) + 8 chunks of h (K=1024)
#define NBLK    64           // persistent grid size (Hn/16 n-tiles)

__device__ __forceinline__ ushort_t f32_to_bf16(float f) {
    union { float f; unsigned int u; } x; x.f = f;
    unsigned int lsb = (x.u >> 16) & 1u;
    unsigned int r = x.u + 0x7FFFu + lsb;
    return (ushort_t)(r >> 16);
}

__device__ __forceinline__ float sigmoidf_(float x) {
    return 1.0f / (1.0f + __expf(-x));
}

union AFrag { v16bf v; int4 q[2]; };

// ---------------------------------------------------------------------------
__global__ void tlstm_cvt_x(const float* __restrict__ x, ushort_t* __restrict__ xb, int n) {
    int i = blockIdx.x * blockDim.x + threadIdx.x;
    if (i < n) xb[i] = f32_to_bf16(x[i]);
}

// ---------------------------------------------------------------------------
// Expand tessarine weight (4, din/4, dout/4) fp32 -> transposed big bf16 [dout][din]
__global__ void tlstm_expand_w(const float* __restrict__ src, ushort_t* __restrict__ dst,
                               int din, int dout) {
    int idx = blockIdx.x * blockDim.x + threadIdx.x;
    int total = din * dout;
    if (idx >= total) return;
    int k = idx % din;
    int n = idx / din;
    int dq_in = din >> 2, dq_out = dout >> 2;
    int r = k / dq_in,  a = k % dq_in;
    int c = n / dq_out, b = n % dq_out;
    const int   compT[16] = {0,1,2,3,  1,0,3,2,  2,3,0,1,  3,2,1,0};
    const float signT[16] = {1.f,1.f,1.f,1.f, -1.f,1.f,-1.f,1.f,
                             1.f,1.f,1.f,1.f, -1.f,1.f,-1.f,1.f};
    int   ci = compT[r*4 + c];
    float s  = signT[r*4 + c];
    float v  = s * src[(size_t)ci * dq_in * dq_out + (size_t)a * dq_out + b];
    dst[(size_t)n * din + k] = f32_to_bf16(v);
}

// ---------------------------------------------------------------------------
__global__ void tlstm_zero(unsigned int* __restrict__ p, int n) {
    int i = blockIdx.x * blockDim.x + threadIdx.x;
    if (i < n) p[i] = 0u;
}

// ---------------------------------------------------------------------------
// Persistent recurrent kernel: grid = NBLK blocks (one 16-col n-tile each),
// block = 128 threads (4 waves = 4 gates). Internal t-loop over all 128 steps
// with a device-wide atomic barrier between steps (all blocks co-resident).
__global__ __launch_bounds__(128)
void tlstm_run(const ushort_t* __restrict__ xb,     // [Tn][Bn][Fn] bf16
               const ushort_t* __restrict__ WxT,    // [4][Hn][Fn] bf16 (transposed)
               const ushort_t* __restrict__ WhT,    // [4][Hn][Hn] bf16 (transposed)
               const float* __restrict__ bf, const float* __restrict__ bi,
               const float* __restrict__ bo, const float* __restrict__ bc,
               ushort_t* __restrict__ hb0,          // [Bn][Hn] bf16 (ping)
               ushort_t* __restrict__ hb1,          // [Bn][Hn] bf16 (pong)
               float* __restrict__ c_state,         // [Bn][Hn]
               float* __restrict__ h_f32,           // [Bn][Hn] (final h)
               unsigned int* __restrict__ ctr)      // device barrier counter
{
    __shared__ ushort_t ldsA[2][Bn][CHUNK + LDSPAD]; // 2 x 17KB staging buffers
    __shared__ float    ldsO[4][4][16][16];          // 16KB gate-output exchange

    const int tid  = threadIdx.x;
    const int lane = tid & 31;
    const int g    = tid >> 5;      // gate: 0=f 1=i 2=o 3=c
    const int half = lane >> 4;
    const int l16  = lane & 15;
    const int n0   = blockIdx.x * 16;

    const float* bias = (g == 0) ? bf : (g == 1) ? bi : (g == 2) ? bo : bc;
    const float bval = bias[n0 + l16];

    const ushort_t* brX = WxT + (size_t)g * Hn * Fn + (size_t)(n0 + l16) * Fn;
    const ushort_t* brH = WhT + (size_t)g * Hn * Hn + (size_t)(n0 + l16) * Hn;

    auto stage = [&](int buf, const ushort_t* __restrict__ src, int ld, int k0) {
        #pragma unroll
        for (int j = 0; j < 8; ++j) {
            int idx = j * 128 + tid;            // 0..1023 int4 slots
            int row = idx >> 4;                 // 16 int4 per row
            int c16 = idx & 15;
            *(int4*)&ldsA[buf][row][c16 * 8] =
                *(const int4*)(src + (size_t)row * ld + k0 + c16 * 8);
        }
    };

    #pragma unroll 1
    for (int t = 0; t < Tn; ++t) {
        const ushort_t* xb_t  = xb + (size_t)t * Bn * Fn;
        const ushort_t* h_in  = (t & 1) ? hb1 : hb0;
        ushort_t*       h_out = (t & 1) ? hb0 : hb1;

        v8f acc0, acc1, acc2, acc3;
        #pragma unroll
        for (int v = 0; v < 8; ++v) { acc0[v] = bval; acc1[v] = bval; acc2[v] = bval; acc3[v] = bval; }

        // Consume one staged chunk: load ALL fragments up front (4 B from L2,
        // 16 A from LDS), then issue 16 WMMAs -> progressive waits, deep
        // ds/vmem pipelining instead of drain-to-zero before each WMMA.
        auto compute_chunk = [&](int buf, const ushort_t* __restrict__ br, int k0) {
#define LOADB(bb, kk)                                                     \
            bb.q[0] = *(const int4*)(br + k0 + (kk) +      half * 8);     \
            bb.q[1] = *(const int4*)(br + k0 + (kk) + 16 + half * 8);
#define LOADA(ff, mt, kk)                                                 \
            ff.q[0] = *(const int4*)&ldsA[buf][(mt)*16 + l16][(kk) +      half * 8]; \
            ff.q[1] = *(const int4*)&ldsA[buf][(mt)*16 + l16][(kk) + 16 + half * 8];
#define WMMA(a_, b_, c_) \
            __builtin_amdgcn_wmma_f32_16x16x32_bf16(false, (a_).v, false, (b_).v, (short)0, (c_), false, false)

            AFrag b0, b1, b2, b3;
            AFrag f00, f10, f20, f30;   // kk = 0,  mt 0..3
            AFrag f01, f11, f21, f31;   // kk = 32
            AFrag f02, f12, f22, f32_;  // kk = 64
            AFrag f03, f13, f23, f33;   // kk = 96

            LOADB(b0,  0) LOADB(b1, 32) LOADB(b2, 64) LOADB(b3, 96)
            LOADA(f00, 0,  0) LOADA(f10, 1,  0) LOADA(f20, 2,  0) LOADA(f30, 3,  0)
            LOADA(f01, 0, 32) LOADA(f11, 1, 32) LOADA(f21, 2, 32) LOADA(f31, 3, 32)
            LOADA(f02, 0, 64) LOADA(f12, 1, 64) LOADA(f22, 2, 64) LOADA(f32_,3, 64)
            LOADA(f03, 0, 96) LOADA(f13, 1, 96) LOADA(f23, 2, 96) LOADA(f33, 3, 96)

            acc0 = WMMA(f00, b0, acc0); acc1 = WMMA(f10, b0, acc1);
            acc2 = WMMA(f20, b0, acc2); acc3 = WMMA(f30, b0, acc3);
            acc0 = WMMA(f01, b1, acc0); acc1 = WMMA(f11, b1, acc1);
            acc2 = WMMA(f21, b1, acc2); acc3 = WMMA(f31, b1, acc3);
            acc0 = WMMA(f02, b2, acc0); acc1 = WMMA(f12, b2, acc1);
            acc2 = WMMA(f22, b2, acc2); acc3 = WMMA(f32_,b2, acc3);
            acc0 = WMMA(f03, b3, acc0); acc1 = WMMA(f13, b3, acc1);
            acc2 = WMMA(f23, b3, acc2); acc3 = WMMA(f33, b3, acc3);
#undef LOADB
#undef LOADA
#undef WMMA
        };

        // Chunk pipeline: cg 0..3 = x (K=512), cg 4..11 = h (K=1024). buf = cg&1.
        stage(0, xb_t, Fn, 0);
        __syncthreads();
        #pragma unroll
        for (int cg = 0; cg < NCHUNK; ++cg) {
            if (cg + 1 < 4)
                stage((cg + 1) & 1, xb_t, Fn, (cg + 1) * CHUNK);
            else if (cg + 1 < NCHUNK)
                stage((cg + 1) & 1, h_in, Hn, (cg + 1 - 4) * CHUNK);
            if (cg < 4) compute_chunk(cg & 1, brX, cg * CHUNK);
            else        compute_chunk(cg & 1, brH, (cg - 4) * CHUNK);
            __syncthreads();
        }

        // D layout: element v -> (M = v + 8*half, N = lane&15)
        #pragma unroll
        for (int v = 0; v < 8; ++v) {
            ldsO[g][0][v + 8 * half][l16] = acc0[v];
            ldsO[g][1][v + 8 * half][l16] = acc1[v];
            ldsO[g][2][v + 8 * half][l16] = acc2[v];
            ldsO[g][3][v + 8 * half][l16] = acc3[v];
        }
        __syncthreads();

        // Fused cell update: 4 m-tiles * 256 elements = 1024, 128 threads
        for (int e = tid; e < 1024; e += 128) {
            int mt = e >> 8;
            int m  = (e >> 4) & 15, n = e & 15;
            float fg = sigmoidf_(ldsO[0][mt][m][n]);
            float ig = sigmoidf_(ldsO[1][mt][m][n]);
            float og = sigmoidf_(ldsO[2][mt][m][n]);
            float ag = ldsO[3][mt][m][n];
            size_t off = (size_t)(mt * 16 + m) * Hn + (n0 + n);
            float cn = ig * tanhf(ag) + fg * c_state[off];
            float hn = og * tanhf(cn);
            c_state[off] = cn;
            h_out[off]   = f32_to_bf16(hn);
            if (t == Tn - 1) h_f32[off] = hn;
        }

        // ---- device-wide barrier between steps ----
        __threadfence();            // make this block's global writes visible (agent)
        __syncthreads();            // all threads' stores fenced before signal
        if (tid == 0) {
            __hip_atomic_fetch_add(ctr, 1u, __ATOMIC_RELEASE, __HIP_MEMORY_SCOPE_AGENT);
            const unsigned int target = (unsigned int)NBLK * (unsigned int)(t + 1);
            while (__hip_atomic_load(ctr, __ATOMIC_ACQUIRE, __HIP_MEMORY_SCOPE_AGENT) < target) {
                __builtin_amdgcn_s_sleep(1);
            }
        }
        __syncthreads();            // release whole block once barrier passed
    }
}

// ---------------------------------------------------------------------------
__global__ void tlstm_final(const float* __restrict__ h, const float* __restrict__ W,
                            const float* __restrict__ b, float* __restrict__ out) {
    int idx = blockIdx.x * blockDim.x + threadIdx.x;
    if (idx >= Bn * Cn) return;
    int cc = idx % Cn, bb = idx / Cn;
    float s = b[cc];
    const float* hr = h + (size_t)bb * Hn;
    for (int k = 0; k < Hn; ++k) s += hr[k] * W[(size_t)k * Cn + cc];
    out[idx] = s;
}

// ---------------------------------------------------------------------------
extern "C" void kernel_launch(void* const* d_in, const int* in_sizes, int n_in,
                              void* d_out, int out_size, void* d_ws, size_t ws_size,
                              hipStream_t stream) {
    (void)in_sizes; (void)n_in; (void)out_size; (void)ws_size;

    const float* x     = (const float*)d_in[0];
    const float* wx[4] = { (const float*)d_in[1], (const float*)d_in[4],
                           (const float*)d_in[7], (const float*)d_in[10] };
    const float* bx[4] = { (const float*)d_in[2], (const float*)d_in[5],
                           (const float*)d_in[8], (const float*)d_in[11] };
    const float* wh[4] = { (const float*)d_in[3], (const float*)d_in[6],
                           (const float*)d_in[9], (const float*)d_in[12] };
    const float* fcoW  = (const float*)d_in[13];
    const float* fcob  = (const float*)d_in[14];
    float* out = (float*)d_out;

    char* ws = (char*)d_ws;
    const size_t XB_OFF   = 0;                                       // T*B*F bf16
    const size_t WXT_OFF  = XB_OFF  + (size_t)Tn * Bn * Fn * 2;      // 4*H*F bf16
    const size_t WHT_OFF  = WXT_OFF + (size_t)4 * Hn * Fn * 2;       // 4*H*H bf16
    const size_t HBF0_OFF = WHT_OFF + (size_t)4 * Hn * Hn * 2;       // B*H bf16
    const size_t HBF1_OFF = HBF0_OFF + (size_t)Bn * Hn * 2;
    const size_t C_OFF    = HBF1_OFF + (size_t)Bn * Hn * 2;          // B*H f32
    const size_t HF_OFF   = C_OFF    + (size_t)Bn * Hn * 4;          // B*H f32
    const size_t CTR_OFF  = HF_OFF   + (size_t)Bn * Hn * 4;          // barrier ctr

    ushort_t*     xb   = (ushort_t*)(ws + XB_OFF);
    ushort_t*     WxT  = (ushort_t*)(ws + WXT_OFF);
    ushort_t*     WhT  = (ushort_t*)(ws + WHT_OFF);
    ushort_t*     hbf0 = (ushort_t*)(ws + HBF0_OFF);
    ushort_t*     hbf1 = (ushort_t*)(ws + HBF1_OFF);
    float*        cst  = (float*)(ws + C_OFF);
    float*        hf   = (float*)(ws + HF_OFF);
    unsigned int* ctr  = (unsigned int*)(ws + CTR_OFF);

    {   // convert x to bf16
        int n = Tn * Bn * Fn;
        tlstm_cvt_x<<<(n + 255) / 256, 256, 0, stream>>>(x, xb, n);
    }
    for (int g = 0; g < 4; ++g) {  // expand input weights
        int n = Fn * Hn;
        tlstm_expand_w<<<(n + 255) / 256, 256, 0, stream>>>(
            wx[g], WxT + (size_t)g * Hn * Fn, Fn, Hn);
    }
    for (int g = 0; g < 4; ++g) {  // expand recurrent weights
        int n = Hn * Hn;
        tlstm_expand_w<<<(n + 255) / 256, 256, 0, stream>>>(
            wh[g], WhT + (size_t)g * Hn * Hn, Hn, Hn);
    }
    {   // zero state region (hbf0, hbf1, c, h_f32, ctr are contiguous)
        int n = (int)(((size_t)Bn * Hn * 2 * 2 + (size_t)Bn * Hn * 4 * 2 + 256) / 4);
        tlstm_zero<<<(n + 255) / 256, 256, 0, stream>>>((unsigned int*)hbf0, n);
    }
    // persistent recurrent kernel: all 128 steps, device-wide barrier per step
    tlstm_run<<<NBLK, 128, 0, stream>>>(xb, WxT, WhT,
                                        bx[0], bx[1], bx[2], bx[3],
                                        hbf0, hbf1, cst, hf, ctr);
    {   // final classifier
        int n = Bn * Cn;
        tlstm_final<<<(n + 255) / 256, 256, 0, stream>>>(hf, fcoW, fcob, out);
    }
}
